// VectorQuantizerEMA_43009802502171
// MI455X (gfx1250) — matrix operand order repaired and checked by reading, not compiled, not run on hardware
//
#include <hip/hip_runtime.h>

// ---------------------------------------------------------------------------
// VQ-VAE EMA vector quantizer for MI455X (gfx1250, wave32, WMMA + async-LDS)
//   z:    (32,512,16,16) f32  -> flat (N=16384, D=256)
//   emb:  (K=8192, D=256) f32
// Outputs (concat, f32): q_st[4194304], loss[1], indices[16384],
//                        new_embedding[2097152], new_ecs[8192], new_ees[2097152]
// ---------------------------------------------------------------------------

#define DDIM    256
#define KEMB    8192
#define NROWS   16384
#define CT_COLS 32                    // columns per staged tile
#define NT      (KEMB / CT_COLS)      // 256 tiles

// output layout (float offsets)
constexpr size_t NQ       = (size_t)NROWS * DDIM;           // 4194304
constexpr size_t LOSS_OFF = NQ;                             // 4194304
constexpr size_t IDX_OFF  = NQ + 1;                         // 4194305
constexpr size_t EMB_OFF  = IDX_OFF + NROWS;                // 4210689
constexpr size_t ECS_OFF  = EMB_OFF + (size_t)KEMB * DDIM;  // 6307841
constexpr size_t EES_OFF  = ECS_OFF + KEMB;                 // 6316033

// workspace layout (bytes)
constexpr size_t WS_IDX   = 0;          // 16384 ints (64 KB)
constexpr size_t WS_ENORM = 65536;      // 8192 floats (32 KB)
constexpr size_t WS_ACC   = 98304;      // 2 floats
constexpr size_t WS_EHI   = 131072;     // 8192*256 bf16 (4 MB)
constexpr size_t WS_ELO   = WS_EHI + (size_t)KEMB * DDIM * 2;   // 4 MB

typedef __attribute__((ext_vector_type(8)))  float          v8f;
typedef __attribute__((ext_vector_type(16))) __bf16         v16bf;
typedef __attribute__((ext_vector_type(16))) unsigned short v16us;
typedef __attribute__((ext_vector_type(8)))  unsigned short v8us;

union FragB { v16us u; v16bf b; };

__device__ __forceinline__ unsigned short f2bf_rne(float f) {
  unsigned u = __float_as_uint(f);
  unsigned r = u + 0x7FFFu + ((u >> 16) & 1u);   // round-to-nearest-even
  return (unsigned short)(r >> 16);
}
__device__ __forceinline__ float bf2f(unsigned short h) {
  return __uint_as_float((unsigned)h << 16);
}

// Async copy of one tile row (256 bf16 = 512B): 32 lanes x 16B each.
// lds_dst is a generic pointer into __shared__; bits[31:0] of the flat
// address are the LDS offset (aperture mapping, ISA 10.2).
__device__ __forceinline__ void stage_rows_async(
    const unsigned short* __restrict__ src, unsigned short* lds_dst,
    int c0, int wsub, int lane) {
#pragma unroll
  for (int o = 0; o < 8; ++o) {
    const int r = wsub * 8 + o;
    unsigned lds_off = (unsigned)(uintptr_t)(lds_dst + r * DDIM + lane * 8);
    unsigned long long ga = (unsigned long long)(uintptr_t)
        (src + ((size_t)(c0 + r) * DDIM + lane * 8));
    asm volatile("global_load_async_to_lds_b128 %0, %1, off"
                 :: "v"(lds_off), "v"(ga) : "memory");
  }
}

// ---------------------------------------------------------------------------
// 1) pack embedding -> bf16 hi/lo planes, and enorm[k] = ||e_k||^2 (one pass)
// ---------------------------------------------------------------------------
__global__ __launch_bounds__(256) void vq_pack_emb(
    const float* __restrict__ emb, unsigned short* __restrict__ ehi,
    unsigned short* __restrict__ elo, float* __restrict__ enorm) {
  const int wid = threadIdx.x >> 5, lane = threadIdx.x & 31;
  const int row = blockIdx.x * 8 + wid;
  const float4* p = (const float4*)(emb + (size_t)row * DDIM + lane * 8);
  float4 a = p[0], b = p[1];
  float v[8] = {a.x, a.y, a.z, a.w, b.x, b.y, b.z, b.w};
  v8us h, l;
  float s = 0.f;
#pragma unroll
  for (int t = 0; t < 8; ++t) {
    unsigned short hh = f2bf_rne(v[t]);
    h[t] = hh;
    l[t] = f2bf_rne(v[t] - bf2f(hh));
    s += v[t] * v[t];
  }
  const size_t off = (size_t)row * DDIM + lane * 8;
  *(v8us*)(ehi + off) = h;
  *(v8us*)(elo + off) = l;
#pragma unroll
  for (int mm = 16; mm >= 1; mm >>= 1) s += __shfl_xor(s, mm);
  if (lane == 0) enorm[row] = s;
}

// ---------------------------------------------------------------------------
// 2) EMA init: new_ees = 0.99*ema_ees, new_ecs = 0.99*ema_ecs; zero accum.
// ---------------------------------------------------------------------------
__global__ __launch_bounds__(256) void vq_init(const float* __restrict__ ecs,
                                               const float* __restrict__ ees,
                                               float* __restrict__ out,
                                               float* __restrict__ acc) {
  const size_t i = (size_t)blockIdx.x * blockDim.x + threadIdx.x;
  const size_t stride = (size_t)gridDim.x * blockDim.x;
  for (size_t j = i; j < (size_t)KEMB * DDIM; j += stride)
    out[EES_OFF + j] = ees[j] * 0.99f;
  for (size_t j = i; j < (size_t)KEMB; j += stride)
    out[ECS_OFF + j] = ecs[j] * 0.99f;
  if (i == 0) { acc[0] = 0.0f; acc[1] = 0.0f; }
}

// ---------------------------------------------------------------------------
// 3) Distance + argmin: split-precision bf16 WMMA, double-buffered async
//    B staging from the packed planes. Block = 8 waves; wave w owns rows
//    [blk*128 + w*16, +16). dist = enorm[k] - 2*dot (||x||^2 const per row).
// ---------------------------------------------------------------------------
__global__ __launch_bounds__(256) void vq_dist_argmin(
    const float* __restrict__ z, const unsigned short* __restrict__ ehi,
    const unsigned short* __restrict__ elo, const float* __restrict__ enorm,
    int* __restrict__ idx) {
  __shared__ __align__(32) unsigned short Bhi[2][CT_COLS * DDIM];  // 2x16 KB
  __shared__ __align__(32) unsigned short Blo[2][CT_COLS * DDIM];  // 2x16 KB

  const int tid  = threadIdx.x;
  const int wid  = tid >> 5;
  const int lane = tid & 31;
  const int m    = lane & 15;   // A row within tile / B-C column within tile
  const int half = lane >> 4;
  const int rowBase = blockIdx.x * 128 + wid * 16;

  // ---- A fragments (registers): 8 K-chunks, hi + lo bf16 ----
  // A lane layout: VGPR0-3 -> K = half*16 + 0..7,
  //                VGPR4-7 -> K = half*16 + 8 + 0..7
  FragB a_hi[8], a_lo[8];
  const float* zrow = z + (size_t)(rowBase + m) * DDIM;
#pragma unroll
  for (int j = 0; j < 8; ++j) {
    const int d0 = j * 32 + half * 16;
#pragma unroll
    for (int t = 0; t < 8; ++t) {
      float f = zrow[d0 + t];
      unsigned short h = f2bf_rne(f);
      a_hi[j].u[t] = h;
      a_lo[j].u[t] = f2bf_rne(f - bf2f(h));
    }
#pragma unroll
    for (int t = 0; t < 8; ++t) {
      float f = zrow[d0 + 8 + t];
      unsigned short h = f2bf_rne(f);
      a_hi[j].u[8 + t] = h;
      a_lo[j].u[8 + t] = f2bf_rne(f - bf2f(h));
    }
  }

  float minv[8];
  int   mini[8];
#pragma unroll
  for (int r = 0; r < 8; ++r) { minv[r] = 3.4e38f; mini[r] = 0; }

  // staging role: waves 0-3 copy hi plane rows, waves 4-7 copy lo plane rows
  const unsigned short* splane = (wid < 4) ? ehi : elo;

  // prologue: stage tile 0 into buffer 0 (8 async ops per wave)
  stage_rows_async(splane, (wid < 4) ? &Bhi[0][0] : &Blo[0][0], 0, wid & 3, lane);

  for (int ct = 0; ct < NT; ++ct) {
    const int buf = ct & 1;
    const int c0  = ct * CT_COLS;

    if (ct + 1 < NT) {
      // issue next tile into the other buffer, then wait for current tile:
      // async loads complete in order, so <=8 outstanding => first 8 landed.
      stage_rows_async(splane, (wid < 4) ? &Bhi[buf ^ 1][0] : &Blo[buf ^ 1][0],
                       (ct + 1) * CT_COLS, wid & 3, lane);
      asm volatile("s_wait_asynccnt 0x8" ::: "memory");
    } else {
      asm volatile("s_wait_asynccnt 0x0" ::: "memory");
    }
    __syncthreads();   // publish this wave's landed LDS data to all waves

#pragma unroll
    for (int sub = 0; sub < 2; ++sub) {
      v8f acc = {0.f, 0.f, 0.f, 0.f, 0.f, 0.f, 0.f, 0.f};
      const unsigned short* bh0 = &Bhi[buf][(sub * 16 + m) * DDIM];
      const unsigned short* bl0 = &Blo[buf][(sub * 16 + m) * DDIM];
#pragma unroll
      for (int j = 0; j < 8; ++j) {
        FragB bhf, blf;   // B lane layout: K = half*16 .. +15 contiguous
        bhf.u = *(const v16us*)(bh0 + j * 32 + half * 16);
        blf.u = *(const v16us*)(bl0 + j * 32 + half * 16);
        acc = __builtin_amdgcn_wmma_f32_16x16x32_bf16(
            false, a_hi[j].b, false, bhf.b, (short)0, acc, false, false);
        acc = __builtin_amdgcn_wmma_f32_16x16x32_bf16(
            false, a_hi[j].b, false, blf.b, (short)0, acc, false, false);
        acc = __builtin_amdgcn_wmma_f32_16x16x32_bf16(
            false, a_lo[j].b, false, bhf.b, (short)0, acc, false, false);
      }
      const int col = c0 + sub * 16 + m;
      const float e2 = enorm[col];
#pragma unroll
      for (int r = 0; r < 8; ++r) {
        float dist = e2 - 2.0f * acc[r];
        if (dist < minv[r]) { minv[r] = dist; mini[r] = col; }
      }
    }
    __syncthreads();   // all waves done reading buf before it is re-staged
  }

  // ---- cross-lane argmin within each 16-lane N-group (xor stays in group) --
#pragma unroll
  for (int r = 0; r < 8; ++r) {
    float v = minv[r]; int ix = mini[r];
#pragma unroll
    for (int mm = 8; mm >= 1; mm >>= 1) {
      float ov = __shfl_xor(v, mm);
      int   oi = __shfl_xor(ix, mm);
      if (ov < v || (ov == v && oi < ix)) { v = ov; ix = oi; }
    }
    if (m == 0) idx[rowBase + r + 8 * half] = ix;  // C layout: M = r + 8*half
  }
}

// ---------------------------------------------------------------------------
// 4) Scatter: q_st, indices, commit-loss partials, EMA accumulation atomics.
//    One block per row n, one thread per d.
// ---------------------------------------------------------------------------
__global__ __launch_bounds__(256) void vq_scatter(
    const float* __restrict__ z, const float* __restrict__ emb,
    const int* __restrict__ idx, float* __restrict__ out,
    float* __restrict__ acc) {
  const int n = blockIdx.x;
  const int d = threadIdx.x;
  const int k = idx[n];
  const float zv = z[(size_t)n * DDIM + d];
  const float q  = emb[(size_t)k * DDIM + d];
  out[(size_t)n * DDIM + d] = zv + (q - zv);          // straight-through
  atomicAdd(&out[EES_OFF + (size_t)k * DDIM + d], 0.01f * zv);   // e_i
  const float diff = q - zv;
  float s = diff * diff;
#pragma unroll
  for (int mm = 16; mm >= 1; mm >>= 1) s += __shfl_xor(s, mm);
  __shared__ float part[8];
  if ((threadIdx.x & 31) == 0) part[threadIdx.x >> 5] = s;
  __syncthreads();
  if (threadIdx.x == 0) {
    float t = 0.f;
#pragma unroll
    for (int i = 0; i < 8; ++i) t += part[i];
    atomicAdd(&acc[0], t);
    out[IDX_OFF + n] = (float)k;
    atomicAdd(&out[ECS_OFF + k], 0.01f);              // n_i
  }
}

// ---------------------------------------------------------------------------
// 5) n_tot = sum(new_ecs); loss = BETA * acc / numel
// ---------------------------------------------------------------------------
__global__ __launch_bounds__(256) void vq_finalize1(float* __restrict__ out,
                                                    float* __restrict__ acc) {
  __shared__ float part[256];
  float s = 0.f;
  for (int i = threadIdx.x; i < KEMB; i += 256) s += out[ECS_OFF + i];
  part[threadIdx.x] = s;
  __syncthreads();
  for (int m = 128; m > 0; m >>= 1) {
    if (threadIdx.x < m) part[threadIdx.x] += part[threadIdx.x + m];
    __syncthreads();
  }
  if (threadIdx.x == 0) {
    acc[1] = part[0];
    out[LOSS_OFF] = 0.25f * acc[0] / (float)NQ;       // BETA * mean sq
  }
}

// ---------------------------------------------------------------------------
// 6) new_embedding[k][d] = new_ees[k][d] / smoothed(k)
// ---------------------------------------------------------------------------
__global__ __launch_bounds__(256) void vq_finalize2(float* __restrict__ out,
                                                    const float* __restrict__ acc) {
  const int k = blockIdx.x;
  const int d = threadIdx.x;
  const float nt = acc[1];
  const float sm = (out[ECS_OFF + k] + 1e-5f) / (nt + (float)KEMB * 1e-5f) * nt;
  out[EMB_OFF + (size_t)k * DDIM + d] =
      out[EES_OFF + (size_t)k * DDIM + d] / sm;
}

// ---------------------------------------------------------------------------
extern "C" void kernel_launch(void* const* d_in, const int* in_sizes, int n_in,
                              void* d_out, int out_size, void* d_ws, size_t ws_size,
                              hipStream_t stream) {
  const float* z   = (const float*)d_in[0];
  const float* emb = (const float*)d_in[1];
  const float* ecs = (const float*)d_in[2];
  const float* ees = (const float*)d_in[3];
  float* out = (float*)d_out;
  char*  ws  = (char*)d_ws;

  int*            ws_idx   = (int*)(ws + WS_IDX);
  float*          ws_enorm = (float*)(ws + WS_ENORM);
  float*          ws_acc   = (float*)(ws + WS_ACC);
  unsigned short* ws_ehi   = (unsigned short*)(ws + WS_EHI);
  unsigned short* ws_elo   = (unsigned short*)(ws + WS_ELO);

  vq_pack_emb   <<<dim3(KEMB / 8),    dim3(256), 0, stream>>>(emb, ws_ehi, ws_elo, ws_enorm);
  vq_init       <<<dim3(2048),        dim3(256), 0, stream>>>(ecs, ees, out, ws_acc);
  vq_dist_argmin<<<dim3(NROWS / 128), dim3(256), 0, stream>>>(z, ws_ehi, ws_elo, ws_enorm, ws_idx);
  vq_scatter    <<<dim3(NROWS),       dim3(256), 0, stream>>>(z, emb, ws_idx, out, ws_acc);
  vq_finalize1  <<<dim3(1),           dim3(256), 0, stream>>>(out, ws_acc);
  vq_finalize2  <<<dim3(KEMB),        dim3(256), 0, stream>>>(out, ws_acc);
}
